// GraphAttentionKAN_1211180778453
// MI455X (gfx1250) — matrix operand order repaired and checked by reading, not compiled, not run on hardware
//
#include <hip/hip_runtime.h>

// ---------------------------------------------------------------------------
// GraphAttentionKAN for MI455X (gfx1250, wave32).
// Dense GEMMs -> v_wmma_f32_16x16x32_bf16 (bf16 in, f32 acc),
//   weights staged block-wide into LDS via GLOBAL_LOAD_ASYNC_TO_LDS_B128.
// Edge attention -> 3-pass atomic segment softmax, bf16 gathers.
// ---------------------------------------------------------------------------

typedef __attribute__((ext_vector_type(16))) __bf16 v16bf;
typedef __attribute__((ext_vector_type(8)))  float  v8f;
typedef unsigned short u16;
typedef unsigned int   u32;
typedef unsigned long long u64;

constexpr int kN   = 50000, kE = 120000, kFN = 64, kEA = 16, kR = 4, kH = 4;
constexpr int kHID = 128, kL = 3, kIDE = 32, kRED = 8, kNG = 64, kNC = 10;
constexpr int kKH  = 128, kEDIM = 24, kFFH = 256, kGK = 8;   // GK = GRID+ORD
constexpr int kGCOLS = 12;                                    // GRID + 2*ORD + 1

// ---------------- scalar helpers ----------------
__device__ __forceinline__ u16 f2bf(float f) {
  u32 u = __float_as_uint(f);
  return (u16)((u + 0x7fffu + ((u >> 16) & 1u)) >> 16);       // RNE
}
__device__ __forceinline__ float bf2f(u16 s) { return __uint_as_float((u32)s << 16); }

// order-preserving float <-> uint for atomicMax
__device__ __forceinline__ u32 ford(float f) {
  u32 u = __float_as_uint(f);
  return (u & 0x80000000u) ? ~u : (u | 0x80000000u);
}
__device__ __forceinline__ float ford_inv(u32 u) {
  return (u & 0x80000000u) ? __uint_as_float(u & 0x7fffffffu) : __uint_as_float(~u);
}
__device__ __forceinline__ float siluf(float x) { return x / (1.f + __expf(-x)); }

__device__ __forceinline__ float wave_sum(float v) {
  for (int o = 16; o > 0; o >>= 1) v += __shfl_down(v, o);
  return __shfl(v, 0);
}

// ---------------- generic fills / converts ----------------
__global__ void k_fill_u32(u32* p, u32 v, long n) {
  long i = (long)blockIdx.x * blockDim.x + threadIdx.x;
  if (i < n) p[i] = v;
}
__global__ void k_f32_to_bf16(const float* __restrict__ s, u16* __restrict__ d, long n) {
  long i = (long)blockIdx.x * blockDim.x + threadIdx.x;
  if (i < n) d[i] = f2bf(s[i]);
}

// ---------------- input assembly ----------------
__global__ void k_build_hcat(const float* __restrict__ x, const int* __restrict__ idt,
                             const float* __restrict__ emb, u16* __restrict__ hcat) {
  long i = (long)blockIdx.x * blockDim.x + threadIdx.x;
  if (i >= (long)kN * 96) return;
  int n = (int)(i / 96), j = (int)(i % 96);
  float v = (j < kFN) ? x[(size_t)n * kFN + j]
                      : emb[(size_t)idt[n] * kIDE + (j - kFN)];
  hcat[i] = f2bf(v);
}

__global__ void k_gate(const float* __restrict__ rel_gate, float* __restrict__ gate) {
  if (blockIdx.x || threadIdx.x) return;
  for (int l = 0; l < kL; ++l) {
    float mx = rel_gate[l * kR];
    for (int r = 1; r < kR; ++r) mx = fmaxf(mx, rel_gate[l * kR + r]);
    float ex[kR], s = 0.f;
    for (int r = 0; r < kR; ++r) { ex[r] = __expf(rel_gate[l * kR + r] - mx); s += ex[r]; }
    for (int r = 0; r < kR; ++r) gate[l * kR + r] = ex[r] / s;
  }
}

// ee_bias[l,r,c] = sum_j lin_edge_w[l,r,c,16+j] * rel_emb[l,r,j]
__global__ void k_eebias(const float* __restrict__ lew, const float* __restrict__ rel_emb,
                         float* __restrict__ eeb) {
  int i = blockIdx.x * blockDim.x + threadIdx.x;
  if (i >= kL * kR * kH * kHID) return;
  int lr = i >> 9, c = i & 511;
  float s = 0.f;
  for (int j = 0; j < kRED; ++j)
    s += lew[((size_t)lr * 512 + c) * kEDIM + kEA + j] * rel_emb[lr * kRED + j];
  eeb[i] = s;
}

__global__ void k_msgbias(const float* __restrict__ conv_bias, const float* __restrict__ gate,
                          int l, float* __restrict__ mb) {
  int c = threadIdx.x;
  if (c >= kHID) return;
  float s = 0.f;
  for (int r = 0; r < kR; ++r)
    s += gate[l * kR + r] * conv_bias[(size_t)(l * kR + r) * kHID + c];
  mb[c] = s;
}

// ---------------- WMMA GEMM:  C(MxN) = act(A(MxK) * Bw(NxK)^T + bias) ----------------
// grid = (ceil(M/16/8), N/16). Block stages the 16xK bf16 weight tile for its
// column block into LDS with async loads; 8 waves compute 8 M-tiles against it.
union FragBF { v16bf v; u16 us[16]; uint4 q[2]; };

__global__ void k_gemm_bf16(const u16* __restrict__ A, const u16* __restrict__ Bw,
                            const float* __restrict__ bias,
                            float* __restrict__ Cf, u16* __restrict__ Cb,
                            int M, int Nn, int K, int do_silu) {
  __shared__ u16 smem[16 * 256];                       // 16 cols x K (K <= 256) = 8KB
  const int tid  = threadIdx.x;
  const int lane = tid & 31;
  const int wave = tid >> 5;
  const int col0 = blockIdx.y << 4;

  // ---- cooperative async stage of B tile: 2*K chunks of 16B ----
  {
    const int kc = K >> 3;                              // 16B chunks per row
    const int nch = 16 * kc;
    for (int t = tid; t < nch; t += 256) {
      const int row = t / kc, cc = t - row * kc;
      const u64 ga = (u64)(Bw + (size_t)(col0 + row) * K + cc * 8);
      const u32 lo = (u32)(size_t)(smem + row * K + cc * 8);
      asm volatile("global_load_async_to_lds_b128 %0, %1, off"
                   :: "v"(lo), "v"(ga) : "memory");
    }
    asm volatile("s_wait_asynccnt 0x0" ::: "memory");
  }
  __syncthreads();

  const int tilesM = M >> 4;
  const int tm = blockIdx.x * 8 + wave;
  if (tm >= tilesM) return;                             // wave-uniform
  const int row0 = tm << 4;
  const int half = lane >> 4, mn = lane & 15;

  v8f acc;
  #pragma unroll
  for (int i = 0; i < 8; ++i) acc[i] = 0.f;

  const u16* arow = A + (size_t)(row0 + mn) * K + half * 8;
  const u16* brow = smem + mn * K + half * 8;           // LDS
  for (int k0 = 0; k0 < K; k0 += 32) {
    FragBF fa, fb;
    fa.q[0] = *(const uint4*)(arow + k0);
    fa.q[1] = *(const uint4*)(arow + k0 + 16);
    fb.q[0] = *(const uint4*)(brow + k0);               // ds_load_b128
    fb.q[1] = *(const uint4*)(brow + k0 + 16);
    if (k0 + 32 < K) __builtin_prefetch(arow + k0 + 32, 0, 0);
    acc = __builtin_amdgcn_wmma_f32_16x16x32_bf16(false, fa.v, false, fb.v,
                                                  (short)0, acc, false, false);
  }

  const int col = col0 + mn;
  const float bv = bias ? bias[col] : 0.f;
  float vals[8];
  #pragma unroll
  for (int v = 0; v < 8; ++v) {
    float xv = acc[v] + bv;
    vals[v] = do_silu ? siluf(xv) : xv;
  }
  #pragma unroll
  for (int v = 0; v < 8; ++v) {
    const size_t idx = (size_t)(row0 + half * 8 + v) * Nn + col;
    if (Cf) Cf[idx] = vals[v];
    if (Cb) Cb[idx] = f2bf(vals[v]);
  }
}

// ---------------- edge attention ----------------
// pass 1: logits + segment max (one wave per edge)
__global__ void k_logit(const u16* __restrict__ xl, const u16* __restrict__ xr,
                        const float* __restrict__ edge_attr, const float* __restrict__ eeb,
                        const float* __restrict__ ew,   // lin_edge_w[l,r]: [512][24] f32
                        const float* __restrict__ attw, // att_w[l,r]: [4][128]
                        const int* __restrict__ src, const int* __restrict__ dst,
                        const int* __restrict__ etype, int r,
                        float* __restrict__ logit, u32* __restrict__ m_u) {
  const int lane = threadIdx.x & 31;
  const int e = blockIdx.x * (blockDim.x >> 5) + (threadIdx.x >> 5);
  if (e >= kE) return;
  if (etype[e] != r) return;
  const int s = src[e], d = dst[e];
  float ea[kEA];
  #pragma unroll
  for (int j = 0; j < kEA; ++j) ea[j] = edge_attr[(size_t)e * kEA + j];
  for (int h = 0; h < kH; ++h) {
    float part = 0.f;
    #pragma unroll
    for (int j = 0; j < 4; ++j) {
      const int c = h * kHID + lane + 32 * j;
      float ve = eeb[c];
      const float* wrow = ew + (size_t)c * kEDIM;
      #pragma unroll
      for (int q = 0; q < kEA; ++q) ve += ea[q] * wrow[q];
      float sfe = bf2f(xl[(size_t)s * 512 + c]) + bf2f(xr[(size_t)d * 512 + c]) + ve;
      sfe = sfe > 0.f ? sfe : 0.2f * sfe;               // leaky_relu(0.2)
      part += sfe * attw[c];
    }
    for (int o = 16; o > 0; o >>= 1) part += __shfl_down(part, o);
    if (lane == 0) {
      logit[(size_t)e * kH + h] = part;
      atomicMax(&m_u[(size_t)d * kH + h], ford(part));
    }
  }
}

// pass 2: a = exp(logit - m[dst]), segment sum
__global__ void k_den(const float* __restrict__ logit, const u32* __restrict__ m_u,
                      const int* __restrict__ dst, const int* __restrict__ etype, int r,
                      float* __restrict__ a, float* __restrict__ den) {
  long i = (long)blockIdx.x * blockDim.x + threadIdx.x;   // e*H + h
  if (i >= (long)kE * kH) return;
  const int e = (int)(i >> 2), h = (int)(i & 3);
  if (etype[e] != r) { a[i] = 0.f; return; }
  const float mv = ford_inv(m_u[(size_t)dst[e] * kH + h]);
  const float av = __expf(logit[i] - mv);
  a[i] = av;
  atomicAdd(&den[(size_t)dst[e] * kH + h], av);
}

// pass 3: h_msg[dst] += gate * mean_h( (a/den) * xl[src] )
__global__ void k_scatter(const float* __restrict__ a, const float* __restrict__ den,
                          const u16* __restrict__ xl,
                          const int* __restrict__ src, const int* __restrict__ dst,
                          const int* __restrict__ etype, int r,
                          const float* __restrict__ gate, int gi,
                          float* __restrict__ hmsg) {
  long i = (long)blockIdx.x * blockDim.x + threadIdx.x;   // e*HID + c
  if (i >= (long)kE * kHID) return;
  const int e = (int)(i >> 7), c = (int)(i & 127);
  if (etype[e] != r) return;
  const int s = src[e], d = dst[e];
  const float g = gate[gi] * (1.f / (float)kH);
  float val = 0.f;
  #pragma unroll
  for (int h = 0; h < kH; ++h) {
    float dv = den[(size_t)d * kH + h];
    dv = (dv == 0.f) ? 1.f : dv;
    val += (a[(size_t)e * kH + h] / dv) * bf2f(xl[(size_t)s * 512 + h * kHID + c]);
  }
  atomicAdd(&hmsg[(size_t)d * kHID + c], g * val);
}

// ---------------- layernorm over 128 channels (one wave per node) ----------------
__global__ void k_ln_node(const float* hin, const float* add, const float* ebias,
                          const float* w, const float* b,
                          float* hout, u16* hb) {
  const int lane = threadIdx.x & 31;
  const int n = blockIdx.x * (blockDim.x >> 5) + (threadIdx.x >> 5);
  if (n >= kN) return;
  float v[4], s = 0.f;
  #pragma unroll
  for (int j = 0; j < 4; ++j) {
    const int c = lane + 32 * j;
    float x = hin[(size_t)n * kHID + c] + add[(size_t)n * kHID + c];
    if (ebias) x += ebias[c];
    v[j] = x; s += x;
  }
  s = wave_sum(s);
  const float mean = s * (1.f / kHID);
  float q = 0.f;
  #pragma unroll
  for (int j = 0; j < 4; ++j) { const float d = v[j] - mean; q += d * d; }
  q = wave_sum(q);
  const float inv = rsqrtf(q * (1.f / kHID) + 1e-5f);
  #pragma unroll
  for (int j = 0; j < 4; ++j) {
    const int c = lane + 32 * j;
    const float y = (v[j] - mean) * inv * w[c] + b[c];
    hout[(size_t)n * kHID + c] = y;
    hb[(size_t)n * kHID + c] = f2bf(y);
  }
}

// ---------------- readout ----------------
__global__ void k_readout_acc(const float* __restrict__ h, const int* __restrict__ batch,
                              float* __restrict__ sum, u32* __restrict__ maxu,
                              float* __restrict__ cnt) {
  long i = (long)blockIdx.x * blockDim.x + threadIdx.x;   // n*HID + c
  if (i >= (long)kN * kHID) return;
  const int n = (int)(i >> 7), c = (int)(i & 127);
  const int g = batch[n];
  const float x = h[i];
  atomicAdd(&sum[(size_t)g * kHID + c], x);
  atomicMax(&maxu[(size_t)g * kHID + c], ford(x));
  if (c == 0) atomicAdd(&cnt[g], 1.f);
}

__global__ void k_readout_fin(const float* __restrict__ sum, const u32* __restrict__ maxu,
                              const float* __restrict__ cnt, float* __restrict__ gvec) {
  int i = blockIdx.x * blockDim.x + threadIdx.x;          // g*256 + c2
  if (i >= kNG * 2 * kHID) return;
  const int g = i >> 8, c2 = i & 255;
  gvec[i] = (c2 < kHID) ? sum[g * kHID + c2] / fmaxf(cnt[g], 1.f)
                        : ford_inv(maxu[g * kHID + (c2 - kHID)]);
}

__global__ void k_ln_group(float* gvec, const float* __restrict__ w, const float* __restrict__ b) {
  __shared__ float red[8];
  const int g = blockIdx.x, t = threadIdx.x;
  const float x = gvec[g * 256 + t];
  float s = x;
  for (int o = 16; o > 0; o >>= 1) s += __shfl_down(s, o);
  if ((t & 31) == 0) red[t >> 5] = s;
  __syncthreads();
  float tot = 0.f;
  for (int j = 0; j < 8; ++j) tot += red[j];
  const float mean = tot * (1.f / 256.f);
  __syncthreads();
  const float d = x - mean;
  float q = d * d;
  for (int o = 16; o > 0; o >>= 1) q += __shfl_down(q, o);
  if ((t & 31) == 0) red[t >> 5] = q;
  __syncthreads();
  float qt = 0.f;
  for (int j = 0; j < 8; ++j) qt += red[j];
  const float inv = rsqrtf(qt * (1.f / 256.f) + 1e-5f);
  gvec[g * 256 + t] = d * inv * w[t] + b[t];
}

// ---------------- KAN ----------------
__device__ __forceinline__ void bsplines8(float x, const float* __restrict__ grid, float* B) {
  float bases[11];
  #pragma unroll
  for (int j = 0; j < 11; ++j) bases[j] = (x >= grid[j] && x < grid[j + 1]) ? 1.f : 0.f;
  #pragma unroll
  for (int k = 1; k <= 3; ++k) {
    #pragma unroll
    for (int j = 0; j < 10; ++j) {
      if (j < 11 - k) {
        bases[j] = (x - grid[j]) / (grid[j + k] - grid[j]) * bases[j]
                 + (grid[j + k + 1] - x) / (grid[j + k + 1] - grid[j + 1]) * bases[j + 1];
      }
    }
  }
  #pragma unroll
  for (int q = 0; q < 8; ++q) B[q] = bases[q];
}

__global__ void k_kan1(const float* __restrict__ gvec, const float* __restrict__ bw,
                       const float* __restrict__ sw, const float* __restrict__ sc,
                       const float* __restrict__ grid, float* __restrict__ z) {
  __shared__ float s_act[256];
  __shared__ float s_B[256 * 8];
  const int g = blockIdx.x, t = threadIdx.x;
  {
    const float x = gvec[g * 256 + t];
    s_act[t] = siluf(x);
    float B8[8];
    bsplines8(x, grid + t * kGCOLS, B8);
    #pragma unroll
    for (int q = 0; q < 8; ++q) s_B[t * 8 + q] = B8[q];
  }
  __syncthreads();
  if (t < kKH) {
    float acc = 0.f;
    for (int i = 0; i < 256; ++i) {
      acc += s_act[i] * bw[(size_t)t * 256 + i];
      float sb = 0.f;
      const float* swp = sw + ((size_t)t * 256 + i) * kGK;
      #pragma unroll
      for (int q = 0; q < 8; ++q) sb += s_B[i * 8 + q] * swp[q];
      acc += sb * sc[(size_t)t * 256 + i];
    }
    z[g * kKH + t] = acc;
  }
}

__global__ void k_kan2(const float* __restrict__ z, const float* __restrict__ bw,
                       const float* __restrict__ sw, const float* __restrict__ sc,
                       const float* __restrict__ grid, float* __restrict__ out) {
  __shared__ float s_act[128];
  __shared__ float s_B[128 * 8];
  const int g = blockIdx.x, t = threadIdx.x;
  {
    const float x = z[g * kKH + t];
    s_act[t] = siluf(x);
    float B8[8];
    bsplines8(x, grid + t * kGCOLS, B8);
    #pragma unroll
    for (int q = 0; q < 8; ++q) s_B[t * 8 + q] = B8[q];
  }
  __syncthreads();
  if (t < kNC) {
    float acc = 0.f;
    for (int i = 0; i < kKH; ++i) {
      acc += s_act[i] * bw[(size_t)t * kKH + i];
      float sb = 0.f;
      const float* swp = sw + ((size_t)t * kKH + i) * kGK;
      #pragma unroll
      for (int q = 0; q < 8; ++q) sb += s_B[i * 8 + q] * swp[q];
      acc += sb * sc[(size_t)t * kKH + i];
    }
    out[g * kNC + t] = acc;
  }
}

// ---------------------------------------------------------------------------
static inline unsigned cdiv(long a, long b) { return (unsigned)((a + b - 1) / b); }

extern "C" void kernel_launch(void* const* d_in, const int* in_sizes, int n_in,
                              void* d_out, int out_size, void* d_ws, size_t ws_size,
                              hipStream_t stream) {
  (void)in_sizes; (void)n_in; (void)out_size; (void)ws_size;
  const float* x          = (const float*)d_in[0];
  const float* edge_attr  = (const float*)d_in[1];
  const int*   id_token   = (const int*)d_in[2];
  const int*   edge_index = (const int*)d_in[3];
  const int*   edge_type  = (const int*)d_in[4];
  const int*   batch      = (const int*)d_in[5];
  const float* id_emb     = (const float*)d_in[6];
  const float* in_proj_w  = (const float*)d_in[7];
  const float* in_proj_b  = (const float*)d_in[8];
  const float* rel_emb    = (const float*)d_in[9];
  const float* lin_l_w    = (const float*)d_in[10];
  const float* lin_l_b    = (const float*)d_in[11];
  const float* lin_r_w    = (const float*)d_in[12];
  const float* lin_r_b    = (const float*)d_in[13];
  const float* lin_edge_w = (const float*)d_in[14];
  const float* att_w      = (const float*)d_in[15];
  const float* conv_bias  = (const float*)d_in[16];
  const float* rel_gate   = (const float*)d_in[17];
  const float* norm1_w    = (const float*)d_in[18];
  const float* norm1_b    = (const float*)d_in[19];
  const float* norm2_w    = (const float*)d_in[20];
  const float* norm2_b    = (const float*)d_in[21];
  const float* ffn_w1     = (const float*)d_in[22];
  const float* ffn_b1     = (const float*)d_in[23];
  const float* ffn_w2     = (const float*)d_in[24];
  const float* ffn_b2     = (const float*)d_in[25];
  const float* ro_norm_w  = (const float*)d_in[26];
  const float* ro_norm_b  = (const float*)d_in[27];
  const float* base_w1    = (const float*)d_in[28];
  const float* spline_w1  = (const float*)d_in[29];
  const float* spline_sc1 = (const float*)d_in[30];
  const float* grid1      = (const float*)d_in[31];
  const float* base_w2    = (const float*)d_in[32];
  const float* spline_w2  = (const float*)d_in[33];
  const float* spline_sc2 = (const float*)d_in[34];
  const float* grid2      = (const float*)d_in[35];

  const int* esrc = edge_index;
  const int* edst = edge_index + kE;

  // ---- workspace carve (256B aligned) ----
  char* base = (char*)d_ws;
  size_t off = 0;
  auto carve = [&](size_t bytes) -> void* {
    off = (off + 255) & ~(size_t)255;
    void* p = base + off;
    off += bytes;
    return p;
  };
  float* h     = (float*)carve((size_t)kN * kHID * 4);
  u16*   hb    = (u16*)  carve((size_t)kN * kHID * 2);
  u16*   hcat  = (u16*)  carve((size_t)kN * 96 * 2);
  u16*   xl    = (u16*)  carve((size_t)kN * 512 * 2);
  u16*   xr    = (u16*)  carve((size_t)kN * 512 * 2);
  u16*   f1    = (u16*)  carve((size_t)kN * kFFH * 2);
  float* f2    = (float*)carve((size_t)kN * kHID * 4);
  float* hmsg  = (float*)carve((size_t)kN * kHID * 4);
  float* logit = (float*)carve((size_t)kE * kH * 4);
  float* aat   = (float*)carve((size_t)kE * kH * 4);
  u32*   m_u   = (u32*)  carve((size_t)kN * kH * 4);
  float* den   = (float*)carve((size_t)kN * kH * 4);
  u16*   w_ip  = (u16*)  carve((size_t)kHID * 96 * 2);
  u16*   w_ll  = (u16*)  carve((size_t)kL * kR * 512 * kHID * 2);
  u16*   w_lr  = (u16*)  carve((size_t)kL * kR * 512 * kHID * 2);
  u16*   w_f1  = (u16*)  carve((size_t)kL * kFFH * kHID * 2);
  u16*   w_f2  = (u16*)  carve((size_t)kL * kHID * kFFH * 2);
  float* eeb   = (float*)carve((size_t)kL * kR * 512 * 4);
  float* gate  = (float*)carve((size_t)kL * kR * 4);
  float* mb    = (float*)carve((size_t)kHID * 4);
  float* rsum  = (float*)carve((size_t)kNG * kHID * 4);
  u32*   rmax  = (u32*)  carve((size_t)kNG * kHID * 4);
  float* rcnt  = (float*)carve((size_t)kNG * 4);
  float* gvec  = (float*)carve((size_t)kNG * 256 * 4);
  float* zbuf  = (float*)carve((size_t)kNG * kKH * 4);

  const int T = 256;
  auto conv = [&](const float* s, u16* d, long n) {
    k_f32_to_bf16<<<cdiv(n, T), T, 0, stream>>>(s, d, n);
  };
  auto fill = [&](void* p, u32 v, long n) {
    k_fill_u32<<<cdiv(n, T), T, 0, stream>>>((u32*)p, v, n);
  };
  auto gemm = [&](const u16* A, const u16* Bw, const float* bias,
                  float* Cf, u16* Cb, int M, int Nn, int K, int silu) {
    dim3 g(cdiv(M >> 4, 8), Nn >> 4);
    k_gemm_bf16<<<g, T, 0, stream>>>(A, Bw, bias, Cf, Cb, M, Nn, K, silu);
  };

  // ---- one-time prep ----
  conv(in_proj_w, w_ip, (long)kHID * 96);
  conv(lin_l_w,  w_ll, (long)kL * kR * 512 * kHID);
  conv(lin_r_w,  w_lr, (long)kL * kR * 512 * kHID);
  conv(ffn_w1,   w_f1, (long)kL * kFFH * kHID);
  conv(ffn_w2,   w_f2, (long)kL * kHID * kFFH);
  k_build_hcat<<<cdiv((long)kN * 96, T), T, 0, stream>>>(x, id_token, id_emb, hcat);
  k_gate<<<1, 32, 0, stream>>>(rel_gate, gate);
  k_eebias<<<cdiv(kL * kR * 512, T), T, 0, stream>>>(lin_edge_w, rel_emb, eeb);

  // ---- input projection: h = silu(hcat @ Wip^T + b) ----
  gemm(hcat, w_ip, in_proj_b, h, hb, kN, kHID, 96, 1);

  // ---- GNN layers ----
  for (int l = 0; l < kL; ++l) {
    fill(hmsg, 0u, (long)kN * kHID);
    for (int r = 0; r < kR; ++r) {
      const int lr = l * kR + r;
      gemm(hb, w_ll + (size_t)lr * 512 * kHID, lin_l_b + (size_t)lr * 512,
           nullptr, xl, kN, 512, kHID, 0);
      gemm(hb, w_lr + (size_t)lr * 512 * kHID, lin_r_b + (size_t)lr * 512,
           nullptr, xr, kN, 512, kHID, 0);
      fill(m_u, 0x00800000u, (long)kN * kH);   // ford(-FLT_MAX)
      fill(den, 0u, (long)kN * kH);
      k_logit<<<cdiv(kE, 8), T, 0, stream>>>(
          xl, xr, edge_attr, eeb + (size_t)lr * 512,
          lin_edge_w + (size_t)lr * 512 * kEDIM, att_w + (size_t)lr * 512,
          esrc, edst, edge_type, r, logit, m_u);
      k_den<<<cdiv((long)kE * kH, T), T, 0, stream>>>(logit, m_u, edst, edge_type, r, aat, den);
      k_scatter<<<cdiv((long)kE * kHID, T), T, 0, stream>>>(
          aat, den, xl, esrc, edst, edge_type, r, gate, lr, hmsg);
    }
    k_msgbias<<<1, kHID, 0, stream>>>(conv_bias, gate, l, mb);
    k_ln_node<<<cdiv(kN, 8), T, 0, stream>>>(h, hmsg, mb,
        norm1_w + (size_t)l * kHID, norm1_b + (size_t)l * kHID, h, hb);
    gemm(hb, w_f1 + (size_t)l * kFFH * kHID, ffn_b1 + (size_t)l * kFFH,
         nullptr, f1, kN, kFFH, kHID, 1);
    gemm(f1, w_f2 + (size_t)l * kHID * kFFH, ffn_b2 + (size_t)l * kHID,
         f2, nullptr, kN, kHID, kFFH, 0);
    k_ln_node<<<cdiv(kN, 8), T, 0, stream>>>(h, f2, nullptr,
        norm2_w + (size_t)l * kHID, norm2_b + (size_t)l * kHID, h, hb);
  }

  // ---- readout + KAN head ----
  fill(rsum, 0u, (long)kNG * kHID);
  fill(rmax, 0x00800000u, (long)kNG * kHID);
  fill(rcnt, 0u, kNG);
  k_readout_acc<<<cdiv((long)kN * kHID, T), T, 0, stream>>>(h, batch, rsum, rmax, rcnt);
  k_readout_fin<<<cdiv(kNG * 256, T), T, 0, stream>>>(rsum, rmax, rcnt, gvec);
  k_ln_group<<<kNG, 256, 0, stream>>>(gvec, ro_norm_w, ro_norm_b);
  k_kan1<<<kNG, 256, 0, stream>>>(gvec, base_w1, spline_w1, spline_sc1, grid1, zbuf);
  k_kan2<<<kNG, 128, 0, stream>>>(zbuf, base_w2, spline_w2, spline_sc2, grid2, (float*)d_out);
}
